// TransformerBlock140B_60610578481405
// MI455X (gfx1250) — matrix operand order repaired and checked by reading, not compiled, not run on hardware
//
#include <hip/hip_runtime.h>
#include <hip/hip_bf16.h>
#include <math.h>

// ---------------- problem constants ----------------
#define B_   2
#define S_   1024
#define D_   2048
#define H_   16
#define F_   8192
#define DH_  128
#define M_   (B_*S_)          // 2048 tokens
#define LAM_  0.1f
#define HS_   0.05f
#define BETA_ 0.3f
#define CLIP_ 10.0f
#define EPS_  1e-6f

typedef __attribute__((ext_vector_type(16))) __bf16 v16bf;
typedef __attribute__((ext_vector_type(8)))  float  v8f;
typedef __attribute__((ext_vector_type(4))) unsigned int v4u;
typedef __attribute__((ext_vector_type(8)))  int v8i;
typedef __attribute__((ext_vector_type(4)))  int v4i;

#if __has_builtin(__builtin_amdgcn_tensor_load_to_lds) && __has_builtin(__builtin_amdgcn_s_wait_tensorcnt)
#define USE_TDM 1
#else
#define USE_TDM 0
#endif

union Frag16 { uint4 q[2]; v16bf v; };

static __device__ inline unsigned short f2bf(float f) {
  union { float f; unsigned int u; } c; c.f = f;
  unsigned int r = c.u + 0x7FFFu + ((c.u >> 16) & 1u);
  return (unsigned short)(r >> 16);
}
static __device__ inline float bf2f(unsigned short h) {
  union { unsigned int u; float f; } c; c.u = ((unsigned int)h) << 16; return c.f;
}
static __device__ inline v8f vzero() {
  v8f z;
#pragma unroll
  for (int i = 0; i < 8; i++) z[i] = 0.0f;
  return z;
}
static __device__ inline v8f wmma_bf16(v16bf a, v16bf b, v8f c) {
  return __builtin_amdgcn_wmma_f32_16x16x32_bf16(false, a, false, b, (short)0, c, false, false);
}

#if USE_TDM
// 2D TDM load: tile_x elements (bf16) per row, tile_y rows; global row stride
// stride_e (elements); optional LDS padding (pad codes per CDNA5 D# group1).
static __device__ inline void tdm_load_2d(unsigned lds_off, unsigned long long gaddr,
                                          unsigned tile_x, unsigned tile_y,
                                          unsigned long long stride_e,
                                          unsigned pad_int_code, unsigned pad_amt_code) {
  v4u g0;
  g0[0] = 1u;                                        // count=1, user descriptor
  g0[1] = lds_off;                                   // D#.lds_addr
  g0[2] = (unsigned)gaddr;                           // D#.global_addr[31:0]
  g0[3] = (unsigned)(gaddr >> 32) | (2u << 30);      // addr[56:32] | type=2
  unsigned w0 = (1u << 16)                           // data_size = 2 bytes
              | (1u << 20)                           // pad_enable
              | (pad_int_code << 22) | (pad_amt_code << 25);
  v8i g1;
  g1[0] = (int)w0;
  g1[1] = (int)(tile_x << 16);                       // tensor_dim0[15:0] @ bit48
  g1[2] = (int)((tile_x >> 16) | (tile_y << 16));    // dim0 hi | tensor_dim1 lo
  g1[3] = (int)((tile_y >> 16) | (tile_x << 16));    // dim1 hi | tile_dim0
  g1[4] = (int)tile_y;                               // tile_dim1 | tile_dim2=0
  g1[5] = (int)(unsigned)stride_e;                   // tensor_dim0_stride lo
  g1[6] = (int)(unsigned)(stride_e >> 32);           // stride hi | dim1_stride lo
  g1[7] = 0;
  v4i z4; z4[0] = z4[1] = z4[2] = z4[3] = 0;
#if __clang_major__ >= 23
  v8i z8;
#pragma unroll
  for (int i = 0; i < 8; i++) z8[i] = 0;
  __builtin_amdgcn_tensor_load_to_lds(g0, g1, z4, z4, z8, 0);
#else
  __builtin_amdgcn_tensor_load_to_lds(g0, g1, z4, z4, 0);
#endif
}
#endif

// ---------------- weight cast + transpose: W[R][C] f32 -> Wt[C][R] bf16 ----------------
__global__ __launch_bounds__(256) void k_wcast(const float* __restrict__ W,
                                               unsigned short* __restrict__ Wt,
                                               int R, int C) {
  __shared__ float t[32][33];
  const int r0 = blockIdx.y * 32, c0 = blockIdx.x * 32;
#pragma unroll
  for (int i = 0; i < 4; i++)
    t[threadIdx.y + i * 8][threadIdx.x] =
        W[(size_t)(r0 + threadIdx.y + i * 8) * C + c0 + threadIdx.x];
  __syncthreads();
#pragma unroll
  for (int i = 0; i < 4; i++)
    Wt[(size_t)(c0 + threadIdx.y + i * 8) * R + r0 + threadIdx.x] =
        f2bf(t[threadIdx.x][threadIdx.y + i * 8]);
}

// ---------------- rmsnorm + cast to bf16 (one block per token row) ----------------
__global__ __launch_bounds__(256) void k_rmsnorm_cast(const float* __restrict__ x,
                                                      const float* __restrict__ g,
                                                      unsigned short* __restrict__ out) {
  __shared__ float red[8];
  const int row = blockIdx.x;
  const float* xr = x + (size_t)row * D_;
  float ss = 0.0f;
#pragma unroll
  for (int i = 0; i < 8; i++) { float v = xr[threadIdx.x + i * 256]; ss += v * v; }
#pragma unroll
  for (int off = 16; off >= 1; off >>= 1) ss += __shfl_xor(ss, off, 32);
  if ((threadIdx.x & 31) == 0) red[threadIdx.x >> 5] = ss;
  __syncthreads();
  float tot = 0.0f;
#pragma unroll
  for (int i = 0; i < 8; i++) tot += red[i];
  const float rs = rsqrtf(tot / (float)D_ + EPS_);
#pragma unroll
  for (int i = 0; i < 8; i++) {
    int c = threadIdx.x + i * 256;
    out[(size_t)row * D_ + c] = f2bf(xr[c] * rs * g[c]);
  }
}

// ---------------- WMMA GEMM: C[M][N] = A_bf16[M][K] * Bt_bf16[K][N] (+epilogue) ----------------
// MODE 0: out f32 = acc ; MODE 1: out f32 = resid + acc ; MODE 2: out bf16 = silu(acc)*aux
// A-tiles staged by TDM (tensor_load_to_lds) when available; double-buffered LDS;
// global_prefetch two K-tiles ahead.
template <int MODE>
__global__ __launch_bounds__(256) void k_gemm_bf16(const unsigned short* __restrict__ A,
                                                   const unsigned short* __restrict__ Bt,
                                                   void* __restrict__ out,
                                                   const float* __restrict__ aux,
                                                   const float* __restrict__ resid,
                                                   int Nn, int Kk) {
  __shared__ __align__(16) unsigned short As[2][128][40];  // [m][k], 80B row stride
  __shared__ __align__(16) unsigned short Bs[2][128][40];  // [n][k]
  const int tid = threadIdx.x;
  const int w = tid >> 5, lane = tid & 31, half = lane >> 4, lr = lane & 15;
  const int wr = w & 3, wc = w >> 2;                 // waves: 4 rows x 2 cols
  const int mbase = blockIdx.y * 128, nbase = blockIdx.x * 128;
  v8f acc[2][4];
#pragma unroll
  for (int i = 0; i < 2; i++)
#pragma unroll
    for (int j = 0; j < 4; j++) acc[i][j] = vzero();

  const int arow = tid >> 1, ahalf = tid & 1;        // A tile loader (fallback): 128 x 32
  const int bk = tid >> 3, bseg = tid & 7;           // B tile loader: 32 x 128 (transposed store)
  const unsigned short* aptr = A + (size_t)(mbase + arow) * Kk + ahalf * 16;
  const unsigned short* bptr = Bt + (size_t)bk * Nn + nbase + bseg * 16;

  uint4 rb0, rb1;  // register staging for next B tile
  rb0 = *(const uint4*)(bptr);
  rb1 = *(const uint4*)(bptr + 8);
  auto commitB = [&](int buf) {
    const unsigned short* h0 = (const unsigned short*)&rb0;
    const unsigned short* h1 = (const unsigned short*)&rb1;
#pragma unroll
    for (int i = 0; i < 8; i++) Bs[buf][bseg * 16 + i][bk] = h0[i];
#pragma unroll
    for (int i = 0; i < 8; i++) Bs[buf][bseg * 16 + 8 + i][bk] = h1[i];
  };
  commitB(0);

#if USE_TDM
  // A tile: 32 elems/row x 128 rows, row stride Kk; pad 64B rows by 16B -> 40-half stride
  if (w == 0)
    tdm_load_2d((unsigned)(size_t)(void*)&As[0][0][0],
                (unsigned long long)(size_t)(A + (size_t)mbase * Kk),
                32u, 128u, (unsigned long long)Kk, 3u, 3u);
#else
  uint4 ra0, ra1;
  ra0 = *(const uint4*)(aptr);
  ra1 = *(const uint4*)(aptr + 8);
  {
    *(uint4*)&As[0][arow][ahalf * 16]     = ra0;
    *(uint4*)&As[0][arow][ahalf * 16 + 8] = ra1;
  }
#endif

  const int nkt = Kk >> 5;
  for (int it = 0; it < nkt; ++it) {
#if USE_TDM
    if (w == 0) __builtin_amdgcn_s_wait_tensorcnt((short)0);
#endif
    __syncthreads();  // LDS[it&1] (A via TDM/copy, B via commit) visible to all waves
    // issue next tile's loads; they overlap the WMMA work below
    if (it + 1 < nkt) {
#if USE_TDM
      if (w == 0)
        tdm_load_2d((unsigned)(size_t)(void*)&As[(it + 1) & 1][0][0],
                    (unsigned long long)(size_t)(A + (size_t)mbase * Kk + (it + 1) * 32),
                    32u, 128u, (unsigned long long)Kk, 3u, 3u);
#else
      const unsigned short* an = aptr + (it + 1) * 32;
      ra0 = *(const uint4*)(an);
      ra1 = *(const uint4*)(an + 8);
#endif
      const unsigned short* bn = bptr + (size_t)(it + 1) * 32 * Nn;
      rb0 = *(const uint4*)(bn);
      rb1 = *(const uint4*)(bn + 8);
    }
    if (it + 2 < nkt) {  // warm L2 two tiles ahead (global_prefetch_b8)
      __builtin_prefetch(aptr + (it + 2) * 32, 0, 3);
      __builtin_prefetch(bptr + (size_t)(it + 2) * 32 * Nn, 0, 3);
    }

    const int buf = it & 1;
    Frag16 af[2], bfr[4];
#pragma unroll
    for (int fm = 0; fm < 2; fm++) {
      const unsigned short* p = &As[buf][wr * 32 + fm * 16 + lr][0];
      af[fm].q[0] = *(const uint4*)(p + half * 8);        // k = half*8 .. +7
      af[fm].q[1] = *(const uint4*)(p + 16 + half * 8);   // k = 16+half*8 .. +7
    }
#pragma unroll
    for (int fn = 0; fn < 4; fn++) {
      const unsigned short* p = &Bs[buf][wc * 64 + fn * 16 + lr][0];
      bfr[fn].q[0] = *(const uint4*)(p + half * 8);       // k = half*8 .. +7
      bfr[fn].q[1] = *(const uint4*)(p + 16 + half * 8);  // k = 16+half*8 .. +7
    }
#pragma unroll
    for (int fm = 0; fm < 2; fm++)
#pragma unroll
      for (int fn = 0; fn < 4; fn++)
        acc[fm][fn] = wmma_bf16(af[fm].v, bfr[fn].v, acc[fm][fn]);

    if (it + 1 < nkt) {
      __syncthreads();          // everyone done reading LDS[(it+1)&1] (used in it-1)
      commitB((it + 1) & 1);
#if !USE_TDM
      *(uint4*)&As[(it + 1) & 1][arow][ahalf * 16]     = ra0;
      *(uint4*)&As[(it + 1) & 1][arow][ahalf * 16 + 8] = ra1;
#endif
    }
  }

  // branch-free epilogue (C layout: lanes 0-15 rows v, lanes 16-31 rows v+8; col = lane%16)
#pragma unroll
  for (int fm = 0; fm < 2; fm++)
#pragma unroll
    for (int fn = 0; fn < 4; fn++)
#pragma unroll
      for (int v = 0; v < 8; v++) {
        int row = mbase + wr * 32 + fm * 16 + half * 8 + v;
        int col = nbase + wc * 64 + fn * 16 + lr;
        size_t idx = (size_t)row * Nn + col;
        float a = acc[fm][fn][v];
        if (MODE == 0) {
          ((float*)out)[idx] = a;
        } else if (MODE == 1) {
          ((float*)out)[idx] = resid[idx] + a;
        } else {
          float u = aux[idx];
          float s = a / (1.0f + __expf(-a));
          ((unsigned short*)out)[idx] = f2bf(s * u);
        }
      }
}

// ---------------- RoPE + bf16 packs (q, q^2, k, k^2, v) ----------------
__global__ __launch_bounds__(256) void k_rope_prep(const float* __restrict__ Qf,
                                                   const float* __restrict__ Kf,
                                                   const float* __restrict__ Vf,
                                                   unsigned short* __restrict__ Qb,
                                                   unsigned short* __restrict__ Q2b,
                                                   unsigned short* __restrict__ Kb,
                                                   unsigned short* __restrict__ K2b,
                                                   unsigned short* __restrict__ Vb) {
  const int tok = blockIdx.x;
  const int spos = tok & (S_ - 1);
  const size_t rb = (size_t)tok * D_;
  const float LOGB = 9.210340371976184f;  // ln(10000)
#pragma unroll
  for (int i = 0; i < 8; i++) {
    int c = threadIdx.x + i * 256;
    int dh = c & 127;
    int d = dh & 63;
    bool hi = dh >= 64;
    float freq = __expf(-(float)d * (LOGB / 64.0f));
    float ang = (float)spos * freq;
    float sn, cs;
    __sincosf(ang, &sn, &cs);
    float q1, q2, k1, k2;
    if (!hi) { q1 = Qf[rb + c]; q2 = Qf[rb + c + 64]; k1 = Kf[rb + c]; k2 = Kf[rb + c + 64]; }
    else     { q1 = Qf[rb + c - 64]; q2 = Qf[rb + c]; k1 = Kf[rb + c - 64]; k2 = Kf[rb + c]; }
    float qr = hi ? (q1 * sn + q2 * cs) : (q1 * cs - q2 * sn);
    float kr = hi ? (k1 * sn + k2 * cs) : (k1 * cs - k2 * sn);
    Qb[rb + c]  = f2bf(qr);  Q2b[rb + c] = f2bf(qr * qr);
    Kb[rb + c]  = f2bf(kr);  K2b[rb + c] = f2bf(kr * kr);
    Vb[rb + c]  = f2bf(Vf[rb + c]);
  }
}

// ---------------- flash attention (causal, s = (qk + LAM*q2k2)/sqrt(DH)) ----------------
// grid: (S/128, B*H), block 256 = 8 waves; wave owns 16 q rows
// Second reference pass drops out exactly: rotate_half on BOTH q,k leaves scores unchanged.
__global__ __launch_bounds__(256) void k_attention(const unsigned short* __restrict__ Qb,
                                                   const unsigned short* __restrict__ Q2b,
                                                   const unsigned short* __restrict__ Kb,
                                                   const unsigned short* __restrict__ K2b,
                                                   const unsigned short* __restrict__ Vb,
                                                   float* __restrict__ Of) {
  __shared__ __align__(16) unsigned short Ks[32][136];     // [key][dh]
  __shared__ __align__(16) unsigned short K2s[32][136];
  __shared__ __align__(16) unsigned short Vt[128][40];     // [dh][key] transposed
  __shared__ __align__(16) unsigned short Pw[8][16][40];   // per-wave P, [row][key]
  const int tid = threadIdx.x;
  const int w = tid >> 5, lane = tid & 31, half = lane >> 4, lr = lane & 15;
  const int qbase = blockIdx.x * 128;
  const int b = blockIdx.y >> 4;
  const int h = blockIdx.y & 15;
  const size_t btok = (size_t)b * S_;
  const int qrow0 = qbase + w * 16;
  const float rscale = 0.08838834764831845f;  // 1/sqrt(128)

  // Q A-fragments (rows = lane%16, per ISA A layout)
  Frag16 qa[4], q2a[4];
#pragma unroll
  for (int dc = 0; dc < 4; dc++) {
    size_t gi = (btok + qrow0 + lr) * D_ + h * DH_ + dc * 32 + half * 8;
    qa[dc].q[0]  = *(const uint4*)(Qb + gi);
    qa[dc].q[1]  = *(const uint4*)(Qb + gi + 16);
    q2a[dc].q[0] = *(const uint4*)(Q2b + gi);
    q2a[dc].q[1] = *(const uint4*)(Q2b + gi + 16);
  }

  v8f o[8];
#pragma unroll
  for (int i = 0; i < 8; i++) o[i] = vzero();
  float m[8], l[8];
#pragma unroll
  for (int v = 0; v < 8; v++) { m[v] = -3.0e38f; l[v] = 0.0f; }

  const int ksteps = (qbase >> 5) + 4;
  const int ldk = tid >> 3, seg = tid & 7;
  for (int kt = 0; kt < ksteps; kt++) {
    __syncthreads();
    {
      size_t gi = (btok + kt * 32 + ldk) * D_ + h * DH_ + seg * 16;
#if USE_TDM
      // K and K2 tiles: 128 elems/row x 32 rows, row stride D; 256B rows padded
      // by 16B -> 136-half LDS stride. One DMA each, issued by wave 0.
      if (w == 0) {
        size_t g0 = (btok + kt * 32) * D_ + h * DH_;
        tdm_load_2d((unsigned)(size_t)(void*)&Ks[0][0],
                    (unsigned long long)(size_t)(Kb + g0),
                    128u, 32u, (unsigned long long)D_, 5u, 3u);
        tdm_load_2d((unsigned)(size_t)(void*)&K2s[0][0],
                    (unsigned long long)(size_t)(K2b + g0),
                    128u, 32u, (unsigned long long)D_, 5u, 3u);
      }
#else
      uint4 a0 = *(const uint4*)(Kb + gi);
      uint4 a1 = *(const uint4*)(Kb + gi + 8);
      *(uint4*)&Ks[ldk][seg * 16]     = a0;
      *(uint4*)&Ks[ldk][seg * 16 + 8] = a1;
      uint4 b0 = *(const uint4*)(K2b + gi);
      uint4 b1 = *(const uint4*)(K2b + gi + 8);
      *(uint4*)&K2s[ldk][seg * 16]     = b0;
      *(uint4*)&K2s[ldk][seg * 16 + 8] = b1;
#endif
      uint4 c0 = *(const uint4*)(Vb + gi);
      uint4 c1 = *(const uint4*)(Vb + gi + 8);
      const unsigned short* h0 = (const unsigned short*)&c0;
      const unsigned short* h1 = (const unsigned short*)&c1;
#pragma unroll
      for (int i = 0; i < 8; i++) Vt[seg * 16 + i][ldk] = h0[i];
#pragma unroll
      for (int i = 0; i < 8; i++) Vt[seg * 16 + 8 + i][ldk] = h1[i];
      if (kt + 1 < ksteps) {  // global_prefetch_b8 next K/V tiles
        size_t gn = gi + (size_t)32 * D_;
        __builtin_prefetch(Kb + gn, 0, 3);
        __builtin_prefetch(K2b + gn, 0, 3);
        __builtin_prefetch(Vb + gn, 0, 3);
      }
#if USE_TDM
      if (w == 0) __builtin_amdgcn_s_wait_tensorcnt((short)0);
#endif
    }
    __syncthreads();

    // scores for two 16-key chunks
    v8f sc[2];
#pragma unroll
    for (int nc = 0; nc < 2; nc++) {
      v8f acc = vzero();
#pragma unroll
      for (int dc = 0; dc < 4; dc++) {
        Frag16 bkf;
        const unsigned short* p = &K2s[nc * 16 + lr][dc * 32 + half * 8];
        bkf.q[0] = *(const uint4*)(p);
        bkf.q[1] = *(const uint4*)(p + 16);
        acc = wmma_bf16(q2a[dc].v, bkf.v, acc);
      }
#pragma unroll
      for (int i = 0; i < 8; i++) acc[i] *= LAM_;
#pragma unroll
      for (int dc = 0; dc < 4; dc++) {
        Frag16 bkf;
        const unsigned short* p = &Ks[nc * 16 + lr][dc * 32 + half * 8];
        bkf.q[0] = *(const uint4*)(p);
        bkf.q[1] = *(const uint4*)(p + 16);
        acc = wmma_bf16(qa[dc].v, bkf.v, acc);
      }
      int kg = kt * 32 + nc * 16 + lr;
#pragma unroll
      for (int v = 0; v < 8; v++) {
        int qg = qrow0 + half * 8 + v;
        float s = acc[v] * rscale;
        acc[v] = (kg > qg) ? -1.0e30f : s;
      }
      sc[nc] = acc;
    }

    // online softmax (row reductions across the 16-lane group)
    float corr[8];
#pragma unroll
    for (int v = 0; v < 8; v++) {
      float t = fmaxf(sc[0][v], sc[1][v]);
#pragma unroll
      for (int off = 1; off < 16; off <<= 1) t = fmaxf(t, __shfl_xor(t, off, 32));
      float mn = fmaxf(m[v], t);
      corr[v] = __expf(m[v] - mn);
      m[v] = mn;
      float p0 = __expf(sc[0][v] - mn);
      float p1 = __expf(sc[1][v] - mn);
      float rs = p0 + p1;
#pragma unroll
      for (int off = 1; off < 16; off <<= 1) rs += __shfl_xor(rs, off, 32);
      l[v] = l[v] * corr[v] + rs;
      Pw[w][half * 8 + v][lr]      = f2bf(p0);
      Pw[w][half * 8 + v][16 + lr] = f2bf(p1);
    }
#pragma unroll
    for (int dc8 = 0; dc8 < 8; dc8++)
#pragma unroll
      for (int i = 0; i < 8; i++) o[dc8][i] *= corr[i];

    // P (16x32) as A-fragment, V tiles as B-fragments -> PV
    Frag16 pf;
    {
      const unsigned short* p = &Pw[w][lr][0];
      pf.q[0] = *(const uint4*)(p + half * 8);
      pf.q[1] = *(const uint4*)(p + 16 + half * 8);
    }
#pragma unroll
    for (int dc8 = 0; dc8 < 8; dc8++) {
      Frag16 bv;
      const unsigned short* p = &Vt[dc8 * 16 + lr][half * 8];
      bv.q[0] = *(const uint4*)(p);
      bv.q[1] = *(const uint4*)(p + 16);
      o[dc8] = wmma_bf16(pf.v, bv.v, o[dc8]);
    }
  }

  // normalize and store f32 (gate drops out: o2 == o1 exactly)
#pragma unroll
  for (int v = 0; v < 8; v++) {
    float inv = 1.0f / l[v];
    size_t rowi = (btok + qrow0 + half * 8 + v) * D_ + h * DH_;
#pragma unroll
    for (int dc8 = 0; dc8 < 8; dc8++)
      Of[rowi + dc8 * 16 + lr] = o[dc8][v] * inv;
  }
}

// ---------------- hadamard mix over each head's 128 lanes, cast bf16 ----------------
__global__ __launch_bounds__(256) void k_hadamard_attn(const float* __restrict__ a,
                                                       unsigned short* __restrict__ out) {
  const int row = blockIdx.x;
  const size_t rb = (size_t)row * D_;
#pragma unroll
  for (int i = 0; i < 8; i++) {
    int c = threadIdx.x + i * 256;
    int hb = c & ~127;
    int dh = c & 127;
    float xv = a[rb + c];
    float pv = a[rb + hb + ((dh + 127) & 127)];
    out[rb + c] = f2bf(xv * (1.0f + HS_ * pv));
  }
}

// ---------------- FFN hadamard over F, bf16 -> bf16 ----------------
__global__ __launch_bounds__(256) void k_ffn_act(const unsigned short* __restrict__ hr,
                                                 unsigned short* __restrict__ out) {
  const int row = blockIdx.x;
  const int cb = blockIdx.y * 2048;
  const size_t rb = (size_t)row * F_;
#pragma unroll
  for (int i = 0; i < 8; i++) {
    int c = cb + threadIdx.x + i * 256;
    int prev = (c + F_ - 1) & (F_ - 1);
    float hv = bf2f(hr[rb + c]);
    float pv = bf2f(hr[rb + prev]);
    out[rb + c] = f2bf(hv * (1.0f + HS_ * pv));
  }
}

// ---------------- elementwise combines ----------------
__global__ __launch_bounds__(256) void k_x2(const float* __restrict__ h1,
                                            const float* __restrict__ x,
                                            float* __restrict__ x2) {
  size_t i = (size_t)blockIdx.x * 256 + threadIdx.x;
  x2[i] = (1.0f + BETA_) * h1[i] - BETA_ * x[i];
}
__global__ __launch_bounds__(256) void k_final(const float* __restrict__ h1,
                                               const float* __restrict__ h2,
                                               float* __restrict__ out) {
  size_t i = (size_t)blockIdx.x * 256 + threadIdx.x;
  float v = 0.5f * (h1[i] + h2[i]);   // ALPHA = 0.5
  out[i] = fminf(fmaxf(v, -CLIP_), CLIP_);
}

// ---------------- host: full pipeline ----------------
extern "C" void kernel_launch(void* const* d_in, const int* in_sizes, int n_in,
                              void* d_out, int out_size, void* d_ws, size_t ws_size,
                              hipStream_t stream) {
  (void)in_sizes; (void)n_in; (void)out_size; (void)ws_size;
  const float* x  = (const float*)d_in[0];
  const float* Wq = (const float*)d_in[1];
  const float* Wk = (const float*)d_in[2];
  const float* Wv = (const float*)d_in[3];
  const float* Wo = (const float*)d_in[4];
  // d_in[5] = gate_w: unused — rotate_half on both q and k makes o2 == o1 exactly.
  const float* n1 = (const float*)d_in[6];
  const float* n2 = (const float*)d_in[7];
  const float* n3 = (const float*)d_in[8];
  const float* n4 = (const float*)d_in[9];
  const float* Wg = (const float*)d_in[10];
  const float* Wu = (const float*)d_in[11];
  const float* Wd = (const float*)d_in[12];
  float* out = (float*)d_out;

  char* ws = (char*)d_ws;
  const size_t MB = 1ull << 20;
  auto U16 = [&](size_t mb) { return (unsigned short*)(ws + mb * MB); };
  auto F32 = [&](size_t mb) { return (float*)(ws + mb * MB); };
  // persistent bf16 weights (128 MB — fits MI455X L2 in bf16)
  unsigned short *WqT = U16(0),  *WkT = U16(8),  *WvT = U16(16), *WoT = U16(24);
  unsigned short *WgT = U16(32), *WuT = U16(64), *WdT = U16(96);
  float *h1 = F32(128), *x2 = F32(144), *h2 = F32(160), *xmid = F32(176);
  unsigned short *xnb = U16(192);
  unsigned short *Qb = U16(200), *Q2b = U16(208), *Kb = U16(216), *K2b = U16(224), *Vb = U16(232);
  unsigned short *attnb = U16(240), *hidb = U16(248);
  // phase-aliased region A (64 MB): QKV f32 + attn f32 early; raw hidden bf16 later
  float *Qf = F32(280), *Kf = F32(296), *Vf = F32(312), *attnf = F32(328);
  unsigned short *hraw = U16(280);
  float *upf = F32(344);  // 64 MB -> total ws use: 408 MB

  dim3 tb(256);
  // one-time weight cast+transpose to K-major bf16
  k_wcast<<<dim3(64, 64),  dim3(32, 8), 0, stream>>>(Wq, WqT, D_, D_);
  k_wcast<<<dim3(64, 64),  dim3(32, 8), 0, stream>>>(Wk, WkT, D_, D_);
  k_wcast<<<dim3(64, 64),  dim3(32, 8), 0, stream>>>(Wv, WvT, D_, D_);
  k_wcast<<<dim3(64, 64),  dim3(32, 8), 0, stream>>>(Wo, WoT, D_, D_);
  k_wcast<<<dim3(64, 256), dim3(32, 8), 0, stream>>>(Wg, WgT, F_, D_);
  k_wcast<<<dim3(64, 256), dim3(32, 8), 0, stream>>>(Wu, WuT, F_, D_);
  k_wcast<<<dim3(256, 64), dim3(32, 8), 0, stream>>>(Wd, WdT, D_, F_);

  auto pass = [&](const float* xin, const float* na, const float* nb2, float* xout) {
    k_rmsnorm_cast<<<M_, tb, 0, stream>>>(xin, na, xnb);
    k_gemm_bf16<0><<<dim3(16, 16), tb, 0, stream>>>(xnb, WqT, Qf, nullptr, nullptr, D_, D_);
    k_gemm_bf16<0><<<dim3(16, 16), tb, 0, stream>>>(xnb, WkT, Kf, nullptr, nullptr, D_, D_);
    k_gemm_bf16<0><<<dim3(16, 16), tb, 0, stream>>>(xnb, WvT, Vf, nullptr, nullptr, D_, D_);
    k_rope_prep<<<M_, tb, 0, stream>>>(Qf, Kf, Vf, Qb, Q2b, Kb, K2b, Vb);
    k_attention<<<dim3(8, 32), tb, 0, stream>>>(Qb, Q2b, Kb, K2b, Vb, attnf);
    k_hadamard_attn<<<M_, tb, 0, stream>>>(attnf, attnb);
    k_gemm_bf16<1><<<dim3(16, 16), tb, 0, stream>>>(attnb, WoT, xmid, nullptr, xin, D_, D_);
    k_rmsnorm_cast<<<M_, tb, 0, stream>>>(xmid, nb2, xnb);
    k_gemm_bf16<0><<<dim3(64, 16), tb, 0, stream>>>(xnb, WuT, upf, nullptr, nullptr, F_, D_);
    k_gemm_bf16<2><<<dim3(64, 16), tb, 0, stream>>>(xnb, WgT, hraw, upf, nullptr, F_, D_);
    k_ffn_act<<<dim3(M_, 4), tb, 0, stream>>>(hraw, hidb);
    k_gemm_bf16<1><<<dim3(16, 16), tb, 0, stream>>>(hidb, WdT, xout, nullptr, xmid, D_, F_);
  };

  pass(x, n1, n2, h1);
  k_x2<<<(M_ * D_) / 256, tb, 0, stream>>>(h1, x, x2);
  pass(x2, n3, n4, h2);
  k_final<<<(M_ * D_) / 256, tb, 0, stream>>>(h1, h2, out);
}
